// MultiHeadAttention_18597208392389
// MI455X (gfx1250) — matrix operand order repaired and checked by reading, not compiled, not run on hardware
//
#include <hip/hip_runtime.h>
#include <hip/hip_bf16.h>
#include <math.h>

// ---------------------------------------------------------------------------
// MHA forward for MI455X (gfx1250, wave32, WMMA, 192MB L2).
//   outputs: attn (32,2048,2048) f32 || out (2,2048,1024) f32
// Pipeline:
//   1) QKV projection -> f16 head-major (Q scaled by 1/8; V stored transposed)
//      + one-time LDS-tiled transpose of Wo to f16 (contiguous WMMA B loads)
//   2) FUSED attention: scores + softmax + attn store + attn@V, one pass.
//      attn touches HBM exactly once (537MB write); K/V live in L2.
//   3) out = O @ Wo + bo   (A and B fragments both contiguous 16B loads)
// ---------------------------------------------------------------------------

typedef __attribute__((ext_vector_type(16))) _Float16 v16h;
typedef __attribute__((ext_vector_type(8)))  _Float16 v8h;
typedef __attribute__((ext_vector_type(8)))  float    v8f;

#define HB     32
#define NSEQ   2048
#define DHEAD  64
#define DMODEL 1024

static __device__ __forceinline__ v8f wmma_f16(v16h a, v16h b, v8f c) {
  return __builtin_amdgcn_wmma_f32_16x16x32_f16(false, a, false, b,
                                                (short)0, c, false, false);
}

static __device__ __forceinline__ v16h cat16(v8h lo, v8h hi) {
  v16h r;
#pragma unroll
  for (int i = 0; i < 8; ++i) { r[i] = lo[i]; r[i + 8] = hi[i]; }
  return r;
}

// A fragment (16x32 f16 MxK) from row-major f16, two contiguous 16B loads.
// Lane l<16: row M=l, K={0..15}; lane>=16: K={16..31}.
static __device__ __forceinline__ v16h load_a16(const _Float16* base, int lda,
                                                int row0, int k0, int lane) {
  const _Float16* p = base + (long)(row0 + (lane & 15)) * lda + k0 + ((lane >> 4) * 16);
  return cat16(*(const v8h*)p, *(const v8h*)(p + 8));
}

// B fragment (32x16 f16 KxN) where logical B = M^T (M row-major, ldm cols):
// B[k][n] = M[n][k] -> 32 contiguous bytes per lane.
static __device__ __forceinline__ v16h load_bT16(const _Float16* base, int ldm,
                                                 int k0, int col0, int lane) {
  const _Float16* p = base + (long)(col0 + (lane & 15)) * ldm + k0 + ((lane >> 4) * 16);
  return cat16(*(const v8h*)p, *(const v8h*)(p + 8));
}

// ---------------------------------------------------------------------------
// 1) QKV projection. transposed==0: Yh[(h*2+b)*2048+n][d] (q,k)
//    transposed==1: Yt[(h*2+b)*64 + d][n]                 (v, for fused attn)
// ---------------------------------------------------------------------------
__global__ void mha_qkv_proj(const float* __restrict__ X, const float* __restrict__ W,
                             const float* __restrict__ bias, _Float16* __restrict__ Y,
                             float scale, int transposed) {
  const int wave = blockIdx.x * (blockDim.x >> 5) + (threadIdx.x >> 5);
  const int lane = threadIdx.x & 31;
  const int mt = wave >> 6, nt = wave & 63;
  const int row0 = mt * 16, col0 = nt * 16;
  const int bb = row0 >> 11, nn = row0 & 2047;
  const int h = col0 >> 6, d0 = col0 & 63;

  v8f acc = {};
#pragma unroll
  for (int k0 = 0; k0 < 64; k0 += 32) {
    const float* pa = X + (long)(row0 + (lane & 15)) * 64 + k0 + ((lane >> 4) * 16);
    v16h a;
#pragma unroll
    for (int i = 0; i < 16; ++i) a[i] = (_Float16)pa[i];
    const float* pb = W + (long)(k0 + ((lane >> 4) * 16)) * DMODEL + col0 + (lane & 15);
    v16h bf;
#pragma unroll
    for (int i = 0; i < 16; ++i) bf[i] = (_Float16)pb[(long)i * DMODEL];
    acc = wmma_f16(a, bf, acc);
  }
  const int c = col0 + (lane & 15);
  const float bval = bias[c];
  const int m0 = (lane >> 4) * 8;
  const int hb = h * 2 + bb;
  if (!transposed) {
    const long obase = (long)hb * NSEQ + nn;
#pragma unroll
    for (int r = 0; r < 8; ++r)
      Y[(obase + m0 + r) * DHEAD + d0 + (lane & 15)] = (_Float16)((acc[r] + bval) * scale);
  } else {
    _Float16* p = Y + ((long)hb * DHEAD + d0 + (lane & 15)) * NSEQ + nn + m0;
#pragma unroll
    for (int r = 0; r < 8; ++r) p[r] = (_Float16)((acc[r] + bval) * scale);
  }
}

// One-time Wo transpose + f32->f16: WoT[n][k] = Wo[k][n]. LDS-tiled so both
// the global read and the global write stay coalesced.
__global__ void mha_wo_T(const float* __restrict__ Wo, _Float16* __restrict__ WoT) {
  __shared__ _Float16 tile[32][33];           // +1 pad: no bank conflicts
  const int bx = blockIdx.x & 31;             // col-tile of Wo
  const int by = blockIdx.x >> 5;             // row-tile of Wo
  const int tx = threadIdx.x & 31;
  const int ty = threadIdx.x >> 5;            // 0..7
#pragma unroll
  for (int j = 0; j < 4; ++j)
    tile[ty + j * 8][tx] = (_Float16)Wo[(long)(by * 32 + ty + j * 8) * DMODEL + bx * 32 + tx];
  __syncthreads();
#pragma unroll
  for (int j = 0; j < 4; ++j)
    WoT[(long)(bx * 32 + ty + j * 8) * DMODEL + by * 32 + tx] = tile[tx][ty + j * 8];
}

// ---------------------------------------------------------------------------
// 2) Fused attention. One block (8 waves) owns (hb, 16 query rows).
//    Wave w owns key strip [w*256, w*256+256).
// ---------------------------------------------------------------------------
__global__ void mha_fused_attn(const _Float16* __restrict__ qh,
                               const _Float16* __restrict__ kh,
                               const _Float16* __restrict__ vt,
                               float* __restrict__ attn,
                               _Float16* __restrict__ om) {
  __shared__ char smem[65536];
  const int hb = blockIdx.x >> 7;
  const int mt = blockIdx.x & 127;
  const int h = hb >> 1, bb = hb & 1;
  const int w = threadIdx.x >> 5;
  const int lane = threadIdx.x & 31;
  const int m0 = (lane >> 4) * 8;
  const int cl = lane & 15;
  const int row0 = mt * 16;
  const int key0 = w * 256;

  const _Float16* qb = qh + (long)hb * NSEQ * DHEAD;
  const _Float16* kb = kh + (long)hb * NSEQ * DHEAD;
  const _Float16* vtb = vt + (long)hb * DHEAD * NSEQ;
  float* ab = attn + (long)hb * NSEQ * NSEQ;

  // ---- Phase A: scores strip, 16 tiles of 16x16 -------------------------
  const v16h a0 = load_a16(qb, DHEAD, row0, 0, lane);
  const v16h a1 = load_a16(qb, DHEAD, row0, 32, lane);
  v8f acc[16];
#pragma unroll
  for (int t = 0; t < 16; ++t) {
    const int col0 = key0 + t * 16;
    v8f z = {};
    z = wmma_f16(a0, load_bT16(kb, DHEAD, 0, col0, lane), z);
    acc[t] = wmma_f16(a1, load_bT16(kb, DHEAD, 32, col0, lane), z);
  }

  // ---- Phase B: block-wide softmax statistics ---------------------------
  float* redM = (float*)smem;          // [8 waves][16 rows]
  float* redS = redM + 128;
  float mrow[8], srow[8];
#pragma unroll
  for (int r = 0; r < 8; ++r) {
    float m = acc[0][r];
#pragma unroll
    for (int t = 1; t < 16; ++t) m = fmaxf(m, acc[t][r]);
#pragma unroll
    for (int off = 1; off < 16; off <<= 1) m = fmaxf(m, __shfl_xor(m, off, 32));
    mrow[r] = m;
  }
  if (cl == 0)
#pragma unroll
    for (int r = 0; r < 8; ++r) redM[w * 16 + m0 + r] = mrow[r];
  __syncthreads();
#pragma unroll
  for (int r = 0; r < 8; ++r) {
    float m = redM[m0 + r];
#pragma unroll
    for (int w2 = 1; w2 < 8; ++w2) m = fmaxf(m, redM[w2 * 16 + m0 + r]);
    mrow[r] = m;
    float s = 0.0f;
#pragma unroll
    for (int t = 0; t < 16; ++t) { float e = __expf(acc[t][r] - m); acc[t][r] = e; s += e; }
#pragma unroll
    for (int off = 1; off < 16; off <<= 1) s += __shfl_xor(s, off, 32);
    srow[r] = s;
  }
  if (cl == 0)
#pragma unroll
    for (int r = 0; r < 8; ++r) redS[w * 16 + m0 + r] = srow[r];
  __syncthreads();
#pragma unroll
  for (int r = 0; r < 8; ++r) {
    float s = redS[m0 + r];
#pragma unroll
    for (int w2 = 1; w2 < 8; ++w2) s += redS[w2 * 16 + m0 + r];
    srow[r] = 1.0f / s;
  }
  __syncthreads();   // redM/redS area reused as prob staging below

  // ---- Phase C: emit attn (f32, HBM) + stage f16 probs in LDS -----------
  _Float16* Sw = (_Float16*)smem + (long)w * 4096;   // [16 rows][256 keys]
#pragma unroll
  for (int t = 0; t < 16; ++t) {
    float* gp = ab + (long)(row0 + m0) * NSEQ + key0 + t * 16 + cl;
#pragma unroll
    for (int r = 0; r < 8; ++r) {
      const float p = acc[t][r] * srow[r];
      gp[(long)r * NSEQ] = p;
      Sw[(m0 + r) * 256 + t * 16 + cl] = (_Float16)p;
    }
  }
  // Same-wave LDS RAW is ordered via DScnt; compiler inserts s_wait_dscnt.

  // ---- Phase D: O^T partial over this wave's 256 keys -------------------
  v8f oacc[4];
  v8f z = {};
#pragma unroll
  for (int t = 0; t < 4; ++t) oacc[t] = z;
#pragma unroll
  for (int kb0 = 0; kb0 < 256; kb0 += 32) {
    const _Float16* bp = Sw + cl * 256 + kb0 + ((lane >> 4) * 16);
    const v16h bf = cat16(*(const v8h*)bp, *(const v8h*)(bp + 8));
    __builtin_prefetch((const void*)(vtb + (long)cl * NSEQ + key0 + kb0 + 64), 0, 0);
#pragma unroll
    for (int t = 0; t < 4; ++t) {
      v16h a = load_a16(vtb, NSEQ, t * 16, key0 + kb0, lane);  // V^T tile (L2)
      oacc[t] = wmma_f16(a, bf, oacc[t]);
    }
  }
  __syncthreads();   // all waves done reading their prob strips

  // ---- Phase E: cross-wave reduction of O^T, store merged-head f16 ------
  float* Opart = (float*)smem;        // [8 waves][64 d][16 qrow]
#pragma unroll
  for (int t = 0; t < 4; ++t)
#pragma unroll
    for (int r = 0; r < 8; ++r)
      Opart[w * 1024 + (t * 16 + m0 + r) * 16 + cl] = oacc[t][r];
  __syncthreads();
  const int tid = threadIdx.x;
#pragma unroll
  for (int j = 0; j < 4; ++j) {
    const int e = tid * 4 + j;           // e = d*16 + qrow
    const int d = e >> 4, qr = e & 15;
    float s = 0.0f;
#pragma unroll
    for (int w2 = 0; w2 < 8; ++w2) s += Opart[w2 * 1024 + e];
    om[(long)(bb * NSEQ + row0 + qr) * DMODEL + h * DHEAD + d] = (_Float16)s;
  }
}

// ---------------------------------------------------------------------------
// 3) out = om (4096x1024 f16) @ Wo + bo -> f32, with B from transposed WoT
//    so every fragment load is contiguous 16B.
// ---------------------------------------------------------------------------
__global__ void mha_outproj(const _Float16* __restrict__ om, const _Float16* __restrict__ woT,
                            const float* __restrict__ bo, float* __restrict__ out) {
  const int wave = blockIdx.x * (blockDim.x >> 5) + (threadIdx.x >> 5);
  const int lane = threadIdx.x & 31;
  const int ns = wave & 15;
  const int mt = wave >> 4;
  const int row0 = mt * 16;

  v8f acc[4];
  v8f z = {};
#pragma unroll
  for (int t = 0; t < 4; ++t) acc[t] = z;

  for (int k0 = 0; k0 < DMODEL; k0 += 32) {
    v16h a = load_a16(om, DMODEL, row0, k0, lane);
#pragma unroll
    for (int t = 0; t < 4; ++t)
      acc[t] = wmma_f16(a, load_bT16(woT, DMODEL, k0, (ns * 4 + t) * 16, lane), acc[t]);
  }
  const int m0 = (lane >> 4) * 8;
#pragma unroll
  for (int t = 0; t < 4; ++t) {
    const int col = (ns * 4 + t) * 16 + (lane & 15);
    const float bval = bo[col];
#pragma unroll
    for (int r = 0; r < 8; ++r)
      out[(long)(row0 + m0 + r) * DMODEL + col] = acc[t][r] + bval;
  }
}

// ---------------------------------------------------------------------------
extern "C" void kernel_launch(void* const* d_in, const int* in_sizes, int n_in,
                              void* d_out, int out_size, void* d_ws, size_t ws_size,
                              hipStream_t stream) {
  const float* q  = (const float*)d_in[0];
  const float* k  = (const float*)d_in[1];
  const float* v  = (const float*)d_in[2];
  const float* Wq = (const float*)d_in[3];
  const float* bq = (const float*)d_in[4];
  const float* Wk = (const float*)d_in[5];
  const float* bk = (const float*)d_in[6];
  const float* Wv = (const float*)d_in[7];
  const float* bv = (const float*)d_in[8];
  const float* Wo = (const float*)d_in[9];
  const float* bo = (const float*)d_in[10];

  float* attn = (float*)d_out;                       // (32,2048,2048)
  float* out  = attn + (long)HB * NSEQ * NSEQ;       // (2,2048,1024)

  const long HSZ = (long)HB * NSEQ * DHEAD;          // 4,194,304 elems
  _Float16* qh   = (_Float16*)d_ws;
  _Float16* kh   = qh + HSZ;
  _Float16* vt   = kh + HSZ;                         // head-transposed V
  _Float16* om   = vt + HSZ;                         // 4096 x 1024
  _Float16* woT  = om + (long)4096 * DMODEL;         // 1024 x 1024 (transposed)

  mha_qkv_proj<<<2048, 256, 0, stream>>>(q, Wq, bq, qh, 0.125f, 0);
  mha_qkv_proj<<<2048, 256, 0, stream>>>(k, Wk, bk, kh, 1.0f, 0);
  mha_qkv_proj<<<2048, 256, 0, stream>>>(v, Wv, bv, vt, 1.0f, 1);
  mha_wo_T<<<1024, 256, 0, stream>>>(Wo, woT);

  // fused attention: 32 heads x 128 query blocks, 8 waves each
  mha_fused_attn<<<HB * 128, 256, 0, stream>>>(qh, kh, vt, attn, om);

  mha_outproj<<<512, 256, 0, stream>>>(om, woT, bo, out);
}
